// DNC_62165356642620
// MI455X (gfx1250) — matrix operand order repaired and checked by reading, not compiled, not run on hardware
//
#include <hip/hip_runtime.h>

#define LAYERS 2
#define HD 512
#define GH 2048   // 4*H
#define BB 32
#define TT 1024
#define NWG 32    // workgroups in recurrent kernel

typedef __attribute__((ext_vector_type(16))) __bf16 v16bf;
typedef __attribute__((ext_vector_type(8)))  float  v8f;
typedef __attribute__((ext_vector_type(4)))  int    i4;

union Frag16 { v16bf v; i4 q[2]; };

__device__ __forceinline__ unsigned short f2bf(float f) {
    unsigned u = __float_as_uint(f);
    unsigned r = (u + 0x7FFFu + ((u >> 16) & 1u)) >> 16;  // RNE
    return (unsigned short)r;
}
__device__ __forceinline__ float bf2f(unsigned short s) {
    return __uint_as_float(((unsigned)s) << 16);
}
__device__ __forceinline__ float sigmoidf(float x) {
    return 1.0f / (1.0f + __expf(-x));
}
__device__ __forceinline__ v8f wmma_bf16(const Frag16& a, const Frag16& b, v8f c) {
    return __builtin_amdgcn_wmma_f32_16x16x32_bf16(false, a.v, false, b.v,
                                                   (short)0, c, false, false);
}

// ---------------- conversion / bias kernels ----------------
__global__ void cvt_f32_bf16(const float* __restrict__ src,
                             unsigned short* __restrict__ dst, int n) {
    int i = blockIdx.x * blockDim.x + threadIdx.x;
    if (i < n) dst[i] = f2bf(src[i]);
}

__global__ void bias_sum(const float* __restrict__ a, const float* __restrict__ b,
                         float* __restrict__ o, int n) {
    int i = blockIdx.x * blockDim.x + threadIdx.x;
    if (i < n) o[i] = a[i] + b[i];
}

// ================= big GEMM building blocks (32M x 64N per wave) =============
// Fragment stage: 2 A-tiles (M, M+16) and 4 B-tiles for one 32-wide k-slab.
struct Stage { Frag16 a0, a1, b[4]; };

__device__ __forceinline__ void stage_load(Stage& s,
                                           const unsigned short* __restrict__ arow0,
                                           const unsigned short* __restrict__ arow1,
                                           const unsigned short* __restrict__ wrow,
                                           int k0, int lh) {
    const unsigned short* ap0 = arow0 + k0 + lh * 8;
    s.a0.q[0] = *(const i4*)(ap0);
    s.a0.q[1] = *(const i4*)(ap0 + 16);
    const unsigned short* ap1 = arow1 + k0 + lh * 8;
    s.a1.q[0] = *(const i4*)(ap1);
    s.a1.q[1] = *(const i4*)(ap1 + 16);
#pragma unroll
    for (int q = 0; q < 4; ++q) {
        const unsigned short* bp = wrow + (long)q * 16 * HD + k0 + lh * 8;
        s.b[q].q[0] = *(const i4*)(bp);
        s.b[q].q[1] = *(const i4*)(bp + 16);
    }
}

__device__ __forceinline__ void stage_mma(v8f acc[2][4], const Stage& s) {
#pragma unroll
    for (int q = 0; q < 4; ++q) acc[0][q] = wmma_bf16(s.a0, s.b[q], acc[0][q]);
#pragma unroll
    for (int q = 0; q < 4; ++q) acc[1][q] = wmma_bf16(s.a1, s.b[q], acc[1][q]);
}

// ---------------- pre-gate GEMM: G[r,n] = A[r,:] @ Wi[n,:] + bias[n] --------
// Logical row r = t*32 + b. A element address = (r&31)*sB + (r>>5)*sT + k.
// Each wave computes one 32(M) x 64(N) tile, 2-stage software pipeline.
__global__ void __launch_bounds__(256, 1)
pregate_kernel(const unsigned short* __restrict__ A, long sB, long sT,
               const unsigned short* __restrict__ Wi,   // [2048,512] bf16
               const float* __restrict__ bias,          // [2048]
               unsigned short* __restrict__ G)          // [T*32, 2048] bf16
{
    const int wave = threadIdx.x >> 5;
    const int lane = threadIdx.x & 31;
    const int ln = lane & 15, lh = lane >> 4;
    const int mbase = blockIdx.x * 32;
    const int nbase = (blockIdx.y * 8 + wave) * 64;

    const int r0 = mbase + ln, r1 = mbase + 16 + ln;
    const unsigned short* arow0 = A + (long)(r0 & 31) * sB + (long)(r0 >> 5) * sT;
    const unsigned short* arow1 = A + (long)(r1 & 31) * sB + (long)(r1 >> 5) * sT;
    const unsigned short* wrow  = Wi + (long)(nbase + ln) * HD;

    v8f acc[2][4] = {{v8f{}, v8f{}, v8f{}, v8f{}}, {v8f{}, v8f{}, v8f{}, v8f{}}};
    Stage s0, s1;
    stage_load(s0, arow0, arow1, wrow, 0, lh);
#pragma unroll 2
    for (int k0 = 0; k0 < HD; k0 += 64) {
        stage_load(s1, arow0, arow1, wrow, k0 + 32, lh);
        stage_mma(acc, s0);
        if (k0 + 64 < HD) stage_load(s0, arow0, arow1, wrow, k0 + 64, lh);
        stage_mma(acc, s1);
    }

#pragma unroll
    for (int mt = 0; mt < 2; ++mt)
#pragma unroll
        for (int q = 0; q < 4; ++q) {
            const int n = nbase + q * 16 + ln;
            const float bv = bias[n];
#pragma unroll
            for (int j = 0; j < 8; ++j) {
                const int m = mbase + mt * 16 + j + 8 * lh;  // VGPR j -> M=j+8*(lane>=16)
                G[(long)m * GH + n] = f2bf(acc[mt][q][j] + bv);
            }
        }
}

// ---------------- recurrent kernel (persistent, 32 WGs) ----------------
// WG wg owns h columns [wg*16, wg*16+16); i.e. gate columns g*512 + wg*16 + c.
// W_hh slice (64 rows x 512) is LDS-resident for the whole layer.
struct RecSmem {
    unsigned short wh[64][HD + 8];  // +8 bf16 pad (16B) to dodge bank conflicts
    float gl[4][BB][16];            // gate exchange buffer i/f/g/o
    float cl[BB][16];               // cell state slice
};

__global__ void __launch_bounds__(256, 1)
recurrent_kernel(const unsigned short* __restrict__ G,    // [T*32,2048] bf16 pre-gates
                 const unsigned short* __restrict__ Wh,   // [2048,512] bf16
                 unsigned short* __restrict__ hstate,     // [2][32][512] bf16 (zeroed)
                 unsigned short* __restrict__ hout,       // [T][32][512] bf16
                 unsigned* __restrict__ bar)
{
    __shared__ RecSmem sm;
    const int wg = blockIdx.x;
    const int tid = threadIdx.x;
    const int wave = tid >> 5, lane = tid & 31;
    const int ln = lane & 15, lh = lane >> 4;
    const int gg = wave >> 1;      // gate group 0..3 (i,f,g,o)
    const int mt = wave & 1;       // M tile (batch 0-15 / 16-31)

    // Stage W_hh slice into LDS: 64 rows x 512 bf16 (64 i4-chunks per row)
    for (int idx = tid; idx < 64 * 64; idx += 256) {
        const int rrow = idx >> 6, chunk = idx & 63;
        const long gr = (long)(rrow >> 4) * 512 + wg * 16 + (rrow & 15); // global gate row
        *(i4*)(&sm.wh[rrow][chunk * 8]) = *(const i4*)(Wh + gr * HD + chunk * 8);
    }
    for (int idx = tid; idx < BB * 16; idx += 256) ((float*)sm.cl)[idx] = 0.0f;
    __syncthreads();

    const int nglob = gg * 512 + wg * 16 + ln;   // global gate column for this lane

    for (int t = 0; t < TT; ++t) {
        const int p = t & 1;

        // Init accumulator chain 0 from pre-gates G (chain 1 starts at zero)
        v8f c0, c1 = {};
        {
            const unsigned short* gp = G + ((long)t * 32 + mt * 16 + lh * 8) * GH + nglob;
#pragma unroll
            for (int j = 0; j < 8; ++j) c0[j] = bf2f(gp[(long)j * GH]);
        }

        // c += h_{t-1}[Mtile] @ Wh_slice^T : dual accumulator chains, 2-stage pipe
        const unsigned short* hrow = hstate + (long)p * BB * HD + (long)(mt * 16 + ln) * HD;
        const unsigned short* brow = &sm.wh[gg * 16 + ln][0];
        Frag16 a0, b0, a1, b1;
        {
            const unsigned short* ap = hrow + lh * 8;
            a0.q[0] = *(const i4*)(ap);
            a0.q[1] = *(const i4*)(ap + 16);
            const unsigned short* bp = brow + lh * 8;
            b0.q[0] = *(const i4*)(bp);
            b0.q[1] = *(const i4*)(bp + 16);
        }
#pragma unroll 2
        for (int k0 = 0; k0 < HD; k0 += 64) {
            {
                const unsigned short* ap = hrow + (k0 + 32) + lh * 8;
                a1.q[0] = *(const i4*)(ap);
                a1.q[1] = *(const i4*)(ap + 16);
                const unsigned short* bp = brow + (k0 + 32) + lh * 8;
                b1.q[0] = *(const i4*)(bp);
                b1.q[1] = *(const i4*)(bp + 16);
            }
            c0 = wmma_bf16(a0, b0, c0);
            if (k0 + 64 < HD) {
                const unsigned short* ap = hrow + (k0 + 64) + lh * 8;
                a0.q[0] = *(const i4*)(ap);
                a0.q[1] = *(const i4*)(ap + 16);
                const unsigned short* bp = brow + (k0 + 64) + lh * 8;
                b0.q[0] = *(const i4*)(bp);
                b0.q[1] = *(const i4*)(bp + 16);
            }
            c1 = wmma_bf16(a1, b1, c1);
        }

        // Scatter gate tile to LDS for cross-gate elementwise combine
        {
            float* gdst = &sm.gl[gg][mt * 16 + lh * 8][ln];
#pragma unroll
            for (int j = 0; j < 8; ++j) gdst[j * 16] = c0[j] + c1[j];
        }
        __syncthreads();

        // Elementwise LSTM cell update on 32x16 slice
        for (int cell = tid; cell < BB * 16; cell += 256) {
            const int b = cell >> 4, col = cell & 15;
            const float xi = sm.gl[0][b][col];
            const float xf = sm.gl[1][b][col];
            const float xg = sm.gl[2][b][col];
            const float xo = sm.gl[3][b][col];
            const float cold = sm.cl[b][col];
            const float cn = sigmoidf(xf) * cold + sigmoidf(xi) * tanhf(xg);
            const float hn = sigmoidf(xo) * tanhf(cn);
            sm.cl[b][col] = cn;
            const unsigned short hb = f2bf(hn);
            hstate[(long)(1 - p) * BB * HD + (long)b * HD + wg * 16 + col] = hb;
            hout[((long)t * BB + b) * HD + wg * 16 + col] = hb;
        }

        // Warm next step's pre-gates while we sit in the barrier
        if (t + 1 < TT) {
            const unsigned short* gnext =
                G + ((long)(t + 1) * 32 + mt * 16 + lh * 8) * GH + nglob;
#pragma unroll
            for (int j = 0; j < 8; ++j)
                __builtin_prefetch(gnext + (long)j * GH, 0, 0);
        }

        // Device-scope grid barrier (monotonic counter; bar zeroed per launch)
        __threadfence();
        __syncthreads();
        if (tid == 0) {
            __hip_atomic_fetch_add(bar, 1u, __ATOMIC_RELEASE, __HIP_MEMORY_SCOPE_AGENT);
            const unsigned target = (unsigned)NWG * (unsigned)(t + 1);
            while (__hip_atomic_load(bar, __ATOMIC_ACQUIRE, __HIP_MEMORY_SCOPE_AGENT) < target)
                __builtin_amdgcn_s_sleep(2);
        }
        __syncthreads();
        __threadfence();
    }
}

// ---------------- output projection: out = h @ Wout^T + b_out ----------------
__global__ void __launch_bounds__(256, 1)
proj_kernel(const unsigned short* __restrict__ Hin,   // [T*32,512] bf16 (time-major)
            const unsigned short* __restrict__ Wo,    // [512,512] bf16
            const float* __restrict__ bo,             // [512]
            float* __restrict__ out)                  // [B,T,H] fp32
{
    const int wave = threadIdx.x >> 5;
    const int lane = threadIdx.x & 31;
    const int ln = lane & 15, lh = lane >> 4;
    const int mbase = blockIdx.x * 32;
    const int nbase = wave * 64;

    const unsigned short* arow0 = Hin + (long)(mbase + ln) * HD;
    const unsigned short* arow1 = Hin + (long)(mbase + 16 + ln) * HD;
    const unsigned short* wrow  = Wo + (long)(nbase + ln) * HD;

    v8f acc[2][4] = {{v8f{}, v8f{}, v8f{}, v8f{}}, {v8f{}, v8f{}, v8f{}, v8f{}}};
    Stage s0, s1;
    stage_load(s0, arow0, arow1, wrow, 0, lh);
#pragma unroll 2
    for (int k0 = 0; k0 < HD; k0 += 64) {
        stage_load(s1, arow0, arow1, wrow, k0 + 32, lh);
        stage_mma(acc, s0);
        if (k0 + 64 < HD) stage_load(s0, arow0, arow1, wrow, k0 + 64, lh);
        stage_mma(acc, s1);
    }

#pragma unroll
    for (int mt = 0; mt < 2; ++mt)
#pragma unroll
        for (int q = 0; q < 4; ++q) {
            const int n = nbase + q * 16 + ln;
            const float bv = bo[n];
#pragma unroll
            for (int j = 0; j < 8; ++j) {
                const int m = mbase + mt * 16 + j + 8 * lh;   // logical row = t*32 + b
                const int bIdx = m & 31, tIdx = m >> 5;
                out[((long)bIdx * TT + tIdx) * HD + n] = acc[mt][q][j] + bv;
            }
        }
}

// ---------------- host launcher ----------------
extern "C" void kernel_launch(void* const* d_in, const int* in_sizes, int n_in,
                              void* d_out, int out_size, void* d_ws, size_t ws_size,
                              hipStream_t stream) {
    (void)in_sizes; (void)n_in; (void)out_size; (void)ws_size;
    const float* x    = (const float*)d_in[0];   // [B,T,H]
    const float* Wih  = (const float*)d_in[1];   // [L,4H,H]
    const float* Whh  = (const float*)d_in[2];   // [L,4H,H]
    const float* bih  = (const float*)d_in[3];   // [L,4H]
    const float* bhh  = (const float*)d_in[4];   // [L,4H]
    const float* Wout = (const float*)d_in[5];   // [H,H]
    const float* bout = (const float*)d_in[6];   // [H]
    float* out = (float*)d_out;

    unsigned char* ws = (unsigned char*)d_ws;
    size_t off = 0;
    auto alloc = [&](size_t bytes) -> void* {
        void* p = ws + off;
        off += (bytes + 255) & ~(size_t)255;
        return p;
    };
    unsigned*       bar  = (unsigned*)      alloc(256);
    unsigned short* xb   = (unsigned short*)alloc((size_t)BB * TT * HD * 2);
    unsigned short* WiB  = (unsigned short*)alloc((size_t)LAYERS * GH * HD * 2);
    unsigned short* WhB  = (unsigned short*)alloc((size_t)LAYERS * GH * HD * 2);
    unsigned short* WoB  = (unsigned short*)alloc((size_t)HD * HD * 2);
    float*          bsum = (float*)         alloc((size_t)LAYERS * GH * 4);
    unsigned short* h0   = (unsigned short*)alloc((size_t)TT * BB * HD * 2);
    unsigned short* h1   = (unsigned short*)alloc((size_t)TT * BB * HD * 2);
    unsigned short* hst  = (unsigned short*)alloc((size_t)2 * BB * HD * 2);
    unsigned short* Gbf  = (unsigned short*)alloc((size_t)TT * BB * GH * 2);

    // Pack inputs to bf16; fold biases.
    {
        int n;
        n = BB * TT * HD;       cvt_f32_bf16<<<(n + 255) / 256, 256, 0, stream>>>(x,    xb,  n);
        n = LAYERS * GH * HD;   cvt_f32_bf16<<<(n + 255) / 256, 256, 0, stream>>>(Wih,  WiB, n);
        n = LAYERS * GH * HD;   cvt_f32_bf16<<<(n + 255) / 256, 256, 0, stream>>>(Whh,  WhB, n);
        n = HD * HD;            cvt_f32_bf16<<<(n + 255) / 256, 256, 0, stream>>>(Wout, WoB, n);
        n = LAYERS * GH;        bias_sum    <<<(n + 255) / 256, 256, 0, stream>>>(bih, bhh, bsum, n);
    }

    const dim3 gemmGrid(TT * BB / 32, GH / (8 * 64));  // (1024, 4)

    // Layer 0: A = x in [B,T,H] -> row(t,b) strides: b-stride = T*H, t-stride = H
    pregate_kernel<<<gemmGrid, 256, 0, stream>>>(xb, (long)TT * HD, (long)HD,
                                                 WiB, bsum, Gbf);
    hipMemsetAsync(hst, 0, (size_t)2 * BB * HD * 2, stream);
    hipMemsetAsync(bar, 0, sizeof(unsigned), stream);
    recurrent_kernel<<<NWG, 256, 0, stream>>>(Gbf, WhB, hst, h0, bar);

    // Layer 1: A = h0 in [T,B,H] -> b-stride = H, t-stride = B*H
    pregate_kernel<<<gemmGrid, 256, 0, stream>>>(h0, (long)HD, (long)BB * HD,
                                                 WiB + (size_t)GH * HD,
                                                 bsum + GH, Gbf);
    hipMemsetAsync(hst, 0, (size_t)2 * BB * HD * 2, stream);
    hipMemsetAsync(bar, 0, sizeof(unsigned), stream);
    recurrent_kernel<<<NWG, 256, 0, stream>>>(Gbf, WhB + (size_t)GH * HD, hst, h1, bar);

    // Output projection
    proj_kernel<<<dim3(TT * BB / 32, 1), 256, 0, stream>>>(h1, WoB, bout, out);
}
